// AttnBlock_33767032881450
// MI455X (gfx1250) — matrix operand order repaired and checked by reading, not compiled, not run on hardware
//
#include <hip/hip_runtime.h>
#include <hip/hip_bf16.h>
#include <math.h>
#include <stdint.h>

// ---------------------------------------------------------------------------
// AttnBlock for MI455X (gfx1250, wave32, WMMA).
// GEMMs in f16 via v_wmma_f32_16x16x32_f16 with f32 accumulation; double-
// buffered LDS staging fed by ASYNCcnt-tracked global->LDS copies (inline
// asm: global_load_async_to_lds_b128); everything else f32.
// ---------------------------------------------------------------------------

typedef _Float16 v16h __attribute__((ext_vector_type(16)));
typedef _Float16 v8h  __attribute__((ext_vector_type(8)));
typedef float    v8f  __attribute__((ext_vector_type(8)));

constexpr int Bsz   = 8;
constexpr int Cch   = 512;
constexpr int Hsz   = 2048;
constexpr int NGRP  = 32;
constexpr int CPG   = Cch / NGRP;          // 16 channels per group
constexpr float EPSV = 1e-6f;

// ---- CDNA5 async global->LDS path (inline asm: portable across toolchains)
#define HAVE_ASYNC_LDS 1

#if HAVE_ASYNC_LDS
__device__ __forceinline__ void async_copy16(const void* g, void* l) {
  // Per-lane 16B global -> LDS, tracked by ASYNCcnt.
  // LDS address = low 32 bits of the flat address of a __shared__ object
  uint32_t laddr = (uint32_t)(uintptr_t)l;
  asm volatile("global_load_async_to_lds_b128 %0, %1, off"
               :
               : "v"(laddr), "v"(g)
               : "memory");
}
__device__ __forceinline__ void async_wait0() {
  asm volatile("s_wait_asynccnt 0x0" ::: "memory");
}
#endif

// ---------------------------------------------------------------------------
// f32 -> f16 weight conversion
// ---------------------------------------------------------------------------
__global__ __launch_bounds__(256) void f32_to_f16_kernel(
    const float* __restrict__ in, _Float16* __restrict__ out, int n) {
  int i = blockIdx.x * 256 + threadIdx.x;
  if (i < n) out[i] = (_Float16)in[i];
}

// ---------------------------------------------------------------------------
// GroupNorm stats: one block per (b, g); group data is contiguous CPG*H floats
// ---------------------------------------------------------------------------
__global__ __launch_bounds__(256) void gn_stats_kernel(
    const float* __restrict__ x, float* __restrict__ stats) {
  const int bg = blockIdx.x;
  const float* p = x + (size_t)bg * (CPG * Hsz);
  float s = 0.f, s2 = 0.f;
  for (int i = threadIdx.x; i < CPG * Hsz; i += 256) {
    float v = p[i];
    s += v; s2 += v * v;
  }
  __shared__ float r1[256], r2[256];
  r1[threadIdx.x] = s; r2[threadIdx.x] = s2;
  __syncthreads();
  for (int off = 128; off > 0; off >>= 1) {
    if ((int)threadIdx.x < off) {
      r1[threadIdx.x] += r1[threadIdx.x + off];
      r2[threadIdx.x] += r2[threadIdx.x + off];
    }
    __syncthreads();
  }
  if (threadIdx.x == 0) {
    const float inv_n = 1.0f / (float)(CPG * Hsz);
    float mean = r1[0] * inv_n;
    float var  = r2[0] * inv_n - mean * mean;
    stats[bg * 2 + 0] = mean;
    stats[bg * 2 + 1] = rsqrtf(var + EPSV);
  }
}

// ---------------------------------------------------------------------------
// GroupNorm apply + transpose: x[b][c][h] -> f16 hnT[b][h][c]
// ---------------------------------------------------------------------------
__global__ __launch_bounds__(256) void gn_apply_kernel(
    const float* __restrict__ x, const float* __restrict__ stats,
    const float* __restrict__ gamma, const float* __restrict__ beta,
    _Float16* __restrict__ hnT) {
  __shared__ float tile[32][33];
  const int b  = blockIdx.z;
  const int h0 = blockIdx.x * 32;
  const int c0 = blockIdx.y * 32;
  const int tx = threadIdx.x & 31;
  const int ty = threadIdx.x >> 5;
#pragma unroll
  for (int i = 0; i < 4; i++) {
    int c = c0 + ty + i * 8;
    tile[ty + i * 8][tx] = x[((size_t)b * Cch + c) * Hsz + h0 + tx];
  }
  __syncthreads();
  const int c = c0 + tx;
  const int g = c >> 4;
  const float mean = stats[(b * NGRP + g) * 2 + 0];
  const float rstd = stats[(b * NGRP + g) * 2 + 1];
  const float ga = gamma[c], be = beta[c];
#pragma unroll
  for (int i = 0; i < 4; i++) {
    int h = h0 + ty + i * 8;
    float v = tile[tx][ty + i * 8];
    hnT[((size_t)b * Hsz + h) * Cch + c] = (_Float16)((v - mean) * rstd * ga + be);
  }
}

// ---------------------------------------------------------------------------
// Softmax over 2048-wide rows: f32 scores -> f16 probabilities
// ---------------------------------------------------------------------------
__global__ __launch_bounds__(256) void softmax_kernel(
    const float* __restrict__ S, _Float16* __restrict__ P) {
  const int i = blockIdx.x;
  const float* row = S + (size_t)i * Hsz;
  float v[8];
  float mx = -3.4e38f;
#pragma unroll
  for (int j = 0; j < 8; j++) {
    v[j] = row[threadIdx.x + j * 256];
    mx = fmaxf(mx, v[j]);
  }
  __shared__ float red[256];
  red[threadIdx.x] = mx;
  __syncthreads();
  for (int off = 128; off > 0; off >>= 1) {
    if ((int)threadIdx.x < off)
      red[threadIdx.x] = fmaxf(red[threadIdx.x], red[threadIdx.x + off]);
    __syncthreads();
  }
  mx = red[0];
  __syncthreads();
  float s = 0.f;
#pragma unroll
  for (int j = 0; j < 8; j++) { v[j] = __expf(v[j] - mx); s += v[j]; }
  red[threadIdx.x] = s;
  __syncthreads();
  for (int off = 128; off > 0; off >>= 1) {
    if ((int)threadIdx.x < off) red[threadIdx.x] += red[threadIdx.x + off];
    __syncthreads();
  }
  const float inv = 1.0f / red[0];
#pragma unroll
  for (int j = 0; j < 8; j++)
    P[(size_t)i * Hsz + threadIdx.x + j * 256] = (_Float16)(v[j] * inv);
}

// ---------------------------------------------------------------------------
// WMMA GEMM: D[M,N] = A[M,K] * Bt[N,K]^T (both K-contiguous).
// Block tile 128x128, 8 waves (2Mx4N), wave tile 64x32 (4x2 wmma tiles),
// K-step 32, double-buffered LDS fed by async copies. All dims multiples of
// 128/32: no guards, EXEC stays all-ones around the WMMAs.
// ---------------------------------------------------------------------------
enum { ST_F16 = 0, ST_F16T = 1, ST_F32SC = 2, ST_F32RES = 3 };

template <int MODE>
__global__ __launch_bounds__(256) void gemm_wmma(
    const _Float16* __restrict__ A, long sA,
    const _Float16* __restrict__ Bt, long sB,
    void* __restrict__ Out, long sO,
    const float* __restrict__ bias,
    const float* __restrict__ resid, long sR,
    int M, int N, int K, float scale) {
  __shared__ _Float16 As[2][128 * 32];
  __shared__ _Float16 Bs[2][128 * 32];

  const int bz = blockIdx.z;
  A  += (size_t)bz * sA;
  Bt += (size_t)bz * sB;

  const int bm = blockIdx.y * 128;
  const int bn = blockIdx.x * 128;
  const int tid  = threadIdx.x;
  const int w    = tid >> 5;
  const int lane = tid & 31;
  const int wm   = (w & 1) * 64;
  const int wn   = (w >> 1) * 32;
  const int half = lane >> 4;
  const int l15  = lane & 15;

  // Per-thread staging coords: thread copies rows (tid>>2) and (tid>>2)+64,
  // 16B chunk (tid&3)*8 halves, for both A and Bt tiles.
  const int lrow = tid >> 2;
  const int lcol = (tid & 3) * 8;
  const _Float16* aP0 = A  + (size_t)(bm + lrow) * K + lcol;
  const _Float16* aP1 = aP0 + (size_t)64 * K;
  const _Float16* bP0 = Bt + (size_t)(bn + lrow) * K + lcol;
  const _Float16* bP1 = bP0 + (size_t)64 * K;
  const int so = lrow * 32 + lcol;          // LDS offset in halves
  const int so1 = so + 64 * 32;

  v8f acc[4][2];
#pragma unroll
  for (int mt = 0; mt < 4; mt++)
#pragma unroll
    for (int nt = 0; nt < 2; nt++)
#pragma unroll
      for (int r = 0; r < 8; r++) acc[mt][nt][r] = 0.f;

  // ---- prologue: fill buffer 0 ----
#if HAVE_ASYNC_LDS
  async_copy16(aP0, &As[0][so]);
  async_copy16(aP1, &As[0][so1]);
  async_copy16(bP0, &Bs[0][so]);
  async_copy16(bP1, &Bs[0][so1]);
  async_wait0();
#else
  {
    v8h a0 = *(const v8h*)aP0, a1 = *(const v8h*)aP1;
    v8h b0 = *(const v8h*)bP0, b1 = *(const v8h*)bP1;
    *(v8h*)&As[0][so]  = a0;  *(v8h*)&As[0][so1] = a1;
    *(v8h*)&Bs[0][so]  = b0;  *(v8h*)&Bs[0][so1] = b1;
  }
#endif
  aP0 += 32; aP1 += 32; bP0 += 32; bP1 += 32;

  int buf = 0;
  for (int k0 = 0; k0 < K; k0 += 32) {
    __syncthreads();                         // LDS[buf] published to all waves
    const bool hasNext = (k0 + 32 < K);

#if HAVE_ASYNC_LDS
    if (hasNext) {                           // async fill of LDS[buf^1]
      async_copy16(aP0, &As[buf ^ 1][so]);
      async_copy16(aP1, &As[buf ^ 1][so1]);
      async_copy16(bP0, &Bs[buf ^ 1][so]);
      async_copy16(bP1, &Bs[buf ^ 1][so1]);
      aP0 += 32; aP1 += 32; bP0 += 32; bP1 += 32;
    }
#else
    v8h a0, a1, b0, b1;
    if (hasNext) {                           // issue next-tile global loads
      a0 = *(const v8h*)aP0; a1 = *(const v8h*)aP1;
      b0 = *(const v8h*)bP0; b1 = *(const v8h*)bP1;
      aP0 += 32; aP1 += 32; bP0 += 32; bP1 += 32;
    }
#endif

    // ---- compute on LDS[buf] ----
    union U { v16h v; v8h h[2]; };
    U a[4], b2[2];
#pragma unroll
    for (int nt = 0; nt < 2; nt++) {
      // B fragment (rows of Bt, K-contiguous): K(e) = half*16 + e
      int r = wn + nt * 16 + l15;
      b2[nt].h[0] = *(const v8h*)&Bs[buf][r * 32 + half * 16];
      b2[nt].h[1] = *(const v8h*)&Bs[buf][r * 32 + half * 16 + 8];
    }
#pragma unroll
    for (int mt = 0; mt < 4; mt++) {
      // A fragment: K(e) = half*16 + e
      int r = wm + mt * 16 + l15;
      a[mt].h[0] = *(const v8h*)&As[buf][r * 32 + half * 16];
      a[mt].h[1] = *(const v8h*)&As[buf][r * 32 + half * 16 + 8];
    }
#pragma unroll
    for (int mt = 0; mt < 4; mt++)
#pragma unroll
      for (int nt = 0; nt < 2; nt++)
        acc[mt][nt] = __builtin_amdgcn_wmma_f32_16x16x32_f16(
            false, a[mt].v, false, b2[nt].v, (short)0, acc[mt][nt], false, false);

    if (hasNext) {
#if HAVE_ASYNC_LDS
      async_wait0();                         // LDS[buf^1] complete before barrier
#else
      *(v8h*)&As[buf ^ 1][so]  = a0;  *(v8h*)&As[buf ^ 1][so1] = a1;
      *(v8h*)&Bs[buf ^ 1][so]  = b0;  *(v8h*)&Bs[buf ^ 1][so1] = b1;
#endif
      buf ^= 1;
    }
  }

  // ---- epilogue: D layout vgpr r -> M = r + 8*half, N = lane&15 ----
#pragma unroll
  for (int mt = 0; mt < 4; mt++)
#pragma unroll
    for (int nt = 0; nt < 2; nt++)
#pragma unroll
      for (int r = 0; r < 8; r++) {
        int m = bm + wm + mt * 16 + r + half * 8;
        int n = bn + wn + nt * 16 + l15;
        float v = acc[mt][nt][r];
        if constexpr (MODE == ST_F16) {
          if (bias) v += bias[m];
          ((_Float16*)Out)[(size_t)bz * sO + (size_t)m * N + n] = (_Float16)v;
        } else if constexpr (MODE == ST_F16T) {
          if (bias) v += bias[m];
          ((_Float16*)Out)[(size_t)bz * sO + (size_t)n * M + m] = (_Float16)v;
        } else if constexpr (MODE == ST_F32SC) {
          ((float*)Out)[(size_t)bz * sO + (size_t)m * N + n] = v * scale;
        } else {
          v += bias[m];
          v += resid[(size_t)bz * sR + (size_t)m * N + n];
          ((float*)Out)[(size_t)bz * sO + (size_t)m * N + n] = v;
        }
      }
}

// ---------------------------------------------------------------------------
// Orchestration
// ---------------------------------------------------------------------------
extern "C" void kernel_launch(void* const* d_in, const int* in_sizes, int n_in,
                              void* d_out, int out_size, void* d_ws, size_t ws_size,
                              hipStream_t stream) {
  (void)in_sizes; (void)n_in; (void)out_size; (void)ws_size;

  const float* x     = (const float*)d_in[0];
  const float* gamma = (const float*)d_in[1];
  const float* beta  = (const float*)d_in[2];
  const float* wq    = (const float*)d_in[3];
  const float* bq    = (const float*)d_in[4];
  const float* wk    = (const float*)d_in[5];
  const float* bk    = (const float*)d_in[6];
  const float* wv    = (const float*)d_in[7];
  const float* bv    = (const float*)d_in[8];
  const float* wp    = (const float*)d_in[9];
  const float* bp    = (const float*)d_in[10];
  float* out = (float*)d_out;

  char* ws = (char*)d_ws;
  size_t off = 0;
  auto alloc = [&](size_t sz) -> char* {
    char* p = ws + off;
    off += (sz + 255) & ~(size_t)255;
    return p;
  };
  float*    stats = (float*)   alloc((size_t)Bsz * NGRP * 2 * sizeof(float));
  _Float16* wq16  = (_Float16*)alloc((size_t)Cch * Cch * sizeof(_Float16));
  _Float16* wk16  = (_Float16*)alloc((size_t)Cch * Cch * sizeof(_Float16));
  _Float16* wv16  = (_Float16*)alloc((size_t)Cch * Cch * sizeof(_Float16));
  _Float16* wp16  = (_Float16*)alloc((size_t)Cch * Cch * sizeof(_Float16));
  _Float16* hnT   = (_Float16*)alloc((size_t)Bsz * Hsz * Cch * sizeof(_Float16)); // [b][h][c]
  _Float16* qT    = (_Float16*)alloc((size_t)Bsz * Hsz * Cch * sizeof(_Float16)); // [b][h][c]
  _Float16* kT    = (_Float16*)alloc((size_t)Bsz * Hsz * Cch * sizeof(_Float16)); // [b][h][c]
  _Float16* vbuf  = (_Float16*)alloc((size_t)Bsz * Cch * Hsz * sizeof(_Float16)); // [b][c][h]
  _Float16* attnT = (_Float16*)alloc((size_t)Bsz * Hsz * Cch * sizeof(_Float16)); // [b][i][c]
  float*    score = (float*)   alloc((size_t)Hsz * Hsz * sizeof(float));          // per-batch reuse
  _Float16* prob  = (_Float16*)alloc((size_t)Hsz * Hsz * sizeof(_Float16));       // per-batch reuse

  const long HC = (long)Hsz * Cch;
  const long CH = (long)Cch * Hsz;

  // 0) weights -> f16
  {
    int n = Cch * Cch;
    int g = (n + 255) / 256;
    f32_to_f16_kernel<<<g, 256, 0, stream>>>(wq, wq16, n);
    f32_to_f16_kernel<<<g, 256, 0, stream>>>(wk, wk16, n);
    f32_to_f16_kernel<<<g, 256, 0, stream>>>(wv, wv16, n);
    f32_to_f16_kernel<<<g, 256, 0, stream>>>(wp, wp16, n);
  }

  // 1) GroupNorm
  gn_stats_kernel<<<Bsz * NGRP, 256, 0, stream>>>(x, stats);
  gn_apply_kernel<<<dim3(Hsz / 32, Cch / 32, Bsz), 256, 0, stream>>>(
      x, stats, gamma, beta, hnT);

  // 2) q/k/v projections: M=C, N=H, K=C; Bt = hnT[b]
  gemm_wmma<ST_F16T><<<dim3(Hsz / 128, Cch / 128, Bsz), 256, 0, stream>>>(
      wq16, 0, hnT, HC, qT, HC, bq, nullptr, 0, Cch, Hsz, Cch, 1.f);
  gemm_wmma<ST_F16T><<<dim3(Hsz / 128, Cch / 128, Bsz), 256, 0, stream>>>(
      wk16, 0, hnT, HC, kT, HC, bk, nullptr, 0, Cch, Hsz, Cch, 1.f);
  gemm_wmma<ST_F16><<<dim3(Hsz / 128, Cch / 128, Bsz), 256, 0, stream>>>(
      wv16, 0, hnT, HC, vbuf, CH, bv, nullptr, 0, Cch, Hsz, Cch, 1.f);

  // 3) attention per batch (score/prob buffers reused; stream serializes)
  const float scale = 1.0f / sqrtf((float)Cch);
  for (int b = 0; b < Bsz; b++) {
    const _Float16* qTb = qT + (size_t)b * HC;
    const _Float16* kTb = kT + (size_t)b * HC;
    const _Float16* vb  = vbuf + (size_t)b * CH;
    _Float16* aTb = attnT + (size_t)b * HC;
    gemm_wmma<ST_F32SC><<<dim3(Hsz / 128, Hsz / 128, 1), 256, 0, stream>>>(
        qTb, 0, kTb, 0, score, 0, nullptr, nullptr, 0, Hsz, Hsz, Cch, scale);
    softmax_kernel<<<Hsz, 256, 0, stream>>>(score, prob);
    gemm_wmma<ST_F16T><<<dim3(Hsz / 128, Cch / 128, 1), 256, 0, stream>>>(
        vb, 0, prob, 0, aTb, 0, nullptr, nullptr, 0, Cch, Hsz, Hsz, 1.f);
  }

  // 4) output projection + bias + residual (f32 out)
  gemm_wmma<ST_F32RES><<<dim3(Hsz / 128, Cch / 128, Bsz), 256, 0, stream>>>(
      wp16, 0, attnT, HC, out, CH, bp, x, CH, Cch, Hsz, Cch, 1.f);
}
